// Head_54631984005567
// MI455X (gfx1250) — matrix operand order repaired and checked by reading, not compiled, not run on hardware
//
#include <hip/hip_runtime.h>

#define BATCH  4
#define SEQ    4096
#define CDIM   1024
#define HDIM   128
#define BT     (BATCH*SEQ)
#define KSPLIT 2

// log2(e) / sqrt(HDIM): folded into q at projection time so attention's
// softmax is a bare exp2 of the WMMA score.
#define QSCL  (0.08838834764831845f * 1.4426950408889634f)

typedef __bf16 bf16;
typedef __attribute__((ext_vector_type(16))) __bf16 v16bf;
typedef __attribute__((ext_vector_type(8)))  float  v8f;

union bfcvt { uint4 u[2]; v16bf v; };

// 32 contiguous bytes -> 16 bf16 fragment
__device__ __forceinline__ v16bf load16bf(const bf16* p) {
    bfcvt t;
    t.u[0] = *(const uint4*)(p);
    t.u[1] = *(const uint4*)(p + 8);
    return t.v;
}
// two separate 16-byte groups (A-fragment pattern: K and K+16)
__device__ __forceinline__ v16bf load2x8(const bf16* p0, const bf16* p1) {
    bfcvt t;
    t.u[0] = *(const uint4*)(p0);
    t.u[1] = *(const uint4*)(p1);
    return t.v;
}

// ---------------------------------------------------------------------------
// Kernel 0: weights f32 [C,H] -> bf16 transposed [3][H][C]  (w0=Wq, w1=Wk, w2=Wv)
// ---------------------------------------------------------------------------
__global__ void wconv_kernel(const float* __restrict__ Wq, const float* __restrict__ Wk,
                             const float* __restrict__ Wv, bf16* __restrict__ wt) {
    int idx = blockIdx.x * 256 + threadIdx.x;           // [0, 3*H*C)
    int w   = idx / (HDIM * CDIM);
    int rem = idx - w * (HDIM * CDIM);
    int h   = rem / CDIM;
    int c   = rem - h * CDIM;
    const float* W = (w == 0) ? Wq : (w == 1 ? Wk : Wv);
    wt[idx] = (bf16)W[c * HDIM + h];
}

// ---------------------------------------------------------------------------
// Kernel 1: q/k/v projections.  grid (BT/16, 3), block 64 (2 waves).
// Wave w computes column tiles 4w..4w+3 (4 independent WMMA chains).
// widx 0 -> q (pre-scaled by QSCL), 1 -> k, 2 -> v (stored as vT[b][h][t]).
// ---------------------------------------------------------------------------
__global__ void __launch_bounds__(64) qkv_kernel(const float* __restrict__ x,
                                                 const bf16* __restrict__ wt,
                                                 bf16* __restrict__ qb,
                                                 bf16* __restrict__ kbuf,
                                                 bf16* __restrict__ vt) {
    const int lane = threadIdx.x & 31;
    const int wave = threadIdx.x >> 5;
    const int widx = blockIdx.y;
    const int g0   = blockIdx.x * 16;
    const int n    = lane & 15;
    const int half = lane >> 4;
    const int ct0  = wave * 4;

    const float* arow = x + (size_t)(g0 + n) * CDIM + half * 16;
    const bf16* brow[4];
    #pragma unroll
    for (int c = 0; c < 4; c++)
        brow[c] = wt + ((size_t)widx * HDIM + (ct0 + c) * 16 + n) * CDIM + half * 16;

    v8f acc[4];
    #pragma unroll
    for (int c = 0; c < 4; c++) acc[c] = (v8f){};

    for (int kc = 0; kc < CDIM; kc += 32) {
        float af[16];
        ((float4*)af)[0] = *(const float4*)(arow + kc);
        ((float4*)af)[1] = *(const float4*)(arow + kc + 4);
        ((float4*)af)[2] = *(const float4*)(arow + kc + 8);
        ((float4*)af)[3] = *(const float4*)(arow + kc + 12);
        v16bf B[4];
        #pragma unroll
        for (int c = 0; c < 4; c++) B[c] = load16bf(brow[c] + kc);
        v16bf A;
        #pragma unroll
        for (int i = 0; i < 16; i++) A[i] = (bf16)af[i];
        #pragma unroll
        for (int c = 0; c < 4; c++)
            acc[c] = __builtin_amdgcn_wmma_f32_16x16x32_bf16(false, A, false, B[c], (short)0, acc[c], false, false);
    }

    const float sc = (widx == 0) ? QSCL : 1.0f;   // fold softmax scale into q
    if (widx < 2) {
        bf16* dst = (widx == 0) ? qb : kbuf;
        #pragma unroll
        for (int c = 0; c < 4; c++)
            #pragma unroll
            for (int r = 0; r < 8; r++) {
                int g = g0 + half * 8 + r;
                dst[(size_t)g * HDIM + (ct0 + c) * 16 + n] = (bf16)(acc[c][r] * sc);
            }
    } else {
        #pragma unroll
        for (int c = 0; c < 4; c++)
            #pragma unroll
            for (int r = 0; r < 8; r++) {
                int g  = g0 + half * 8 + r;
                int bb = g >> 12;            // /SEQ
                int t  = g & (SEQ - 1);
                vt[((size_t)bb * HDIM + (ct0 + c) * 16 + n) * SEQ + t] = (bf16)acc[c][r];
            }
    }
}

// ---------------------------------------------------------------------------
// Kernel 2: attention partials. grid (SEQ/64, B, KSPLIT), block 128 (4 waves).
// Each wave owns 16 query rows and the interleaved chunk set j % KSPLIT == s
// (partials over disjoint key sets are additive since there is no max-rescale).
// Row-sum of P comes from a 9th WMMA against a constant ones-column fragment.
// Writes bf16 partial O and f32 partial row-sum l.
// ---------------------------------------------------------------------------
__global__ void __launch_bounds__(128) attn_kernel(const bf16* __restrict__ qb,
                                                   const bf16* __restrict__ kbuf,
                                                   const bf16* __restrict__ vt,
                                                   bf16* __restrict__ pO,
                                                   float* __restrict__ pL) {
    __shared__ bf16 pbuf[4][16 * 40];   // per-wave P tile, 80B row stride (16B aligned)
    const int lane  = threadIdx.x & 31;
    const int wave  = threadIdx.x >> 5;
    const int b     = blockIdx.y;
    const int split = blockIdx.z;
    const int qt    = gridDim.x - 1 - blockIdx.x;       // heavy tiles first
    const int qbase = qt * 64 + wave * 16;
    const int n     = lane & 15;
    const int half  = lane >> 4;

    // Q fragments: 4 x (16x32) A-fragments covering H=128
    v16bf Qf[4];
    {
        const bf16* qrow = qb + (size_t)(b * SEQ + qbase + n) * HDIM;
        #pragma unroll
        for (int hc = 0; hc < 4; hc++) {
            const bf16* p = qrow + hc * 32 + half * 16;
            Qf[hc] = load16bf(p);
        }
    }

    // Constant ones-column B-fragment: column n==0 is 1.0 for every k.
    v16bf onesf;
    {
        bf16 o = (bf16)((n == 0) ? 1.0f : 0.0f);
        #pragma unroll
        for (int i = 0; i < 16; i++) onesf[i] = o;
    }

    v8f acc[9];                                    // [0..7]=O tiles, [8]=row-sum tile
    #pragma unroll
    for (int ct = 0; ct < 9; ct++) acc[ct] = (v8f){};

    const bf16* kbb = kbuf + (size_t)b * SEQ * HDIM;
    const bf16* vtb = vt   + (size_t)b * HDIM * SEQ;
    bf16* pw = pbuf[wave];

    const int jmax = (qbase + 15) >> 5;
    for (int j = split; j <= jmax; j += KSPLIT) {
        const int kb0 = j * 32;
        // batched K fragment loads (one clause), then the S WMMA chains
        v16bf Kf[8];
        #pragma unroll
        for (int hc = 0; hc < 4; hc++) {
            const bf16* krow0 = kbb + (size_t)(kb0 + n) * HDIM + hc * 32 + half * 16;
            Kf[hc * 2]     = load16bf(krow0);
            Kf[hc * 2 + 1] = load16bf(krow0 + (size_t)16 * HDIM);
        }
        v8f S0 = (v8f){}, S1 = (v8f){};
        #pragma unroll
        for (int hc = 0; hc < 4; hc++) {
            S0 = __builtin_amdgcn_wmma_f32_16x16x32_bf16(false, Qf[hc], false, Kf[hc * 2],     (short)0, S0, false, false);
            S1 = __builtin_amdgcn_wmma_f32_16x16x32_bf16(false, Qf[hc], false, Kf[hc * 2 + 1], (short)0, S1, false, false);
        }
        float p0[8], p1[8];
        #pragma unroll
        for (int r = 0; r < 8; r++) { p0[r] = S0[r]; p1[r] = S1[r]; }
        if (kb0 + 31 > qbase) {          // chunk crosses diagonal: causal mask
            #pragma unroll
            for (int r = 0; r < 8; r++) {
                int t = qbase + half * 8 + r;
                if (kb0 + n > t)      p0[r] = -__builtin_inff();
                if (kb0 + 16 + n > t) p1[r] = -__builtin_inff();
            }
        }
        #pragma unroll
        for (int r = 0; r < 8; r++) { p0[r] = exp2f(p0[r]); p1[r] = exp2f(p1[r]); }

        // P (C/D layout) -> LDS -> A-fragment layout
        #pragma unroll
        for (int r = 0; r < 8; r++) {
            int row = half * 8 + r;
            pw[row * 40 + n]      = (bf16)p0[r];
            pw[row * 40 + 16 + n] = (bf16)p1[r];
        }
        v16bf Pf;
        {
            const bf16* pp = pw + n * 40 + half * 16;   // row = lane%16
            Pf = load16bf(pp);
        }
        // batched V fragment loads, then 9 independent WMMA chains
        v16bf Vf[8];
        #pragma unroll
        for (int ct = 0; ct < 8; ct++)
            Vf[ct] = load16bf(vtb + (size_t)(ct * 16 + n) * SEQ + kb0 + half * 16);
        #pragma unroll
        for (int ct = 0; ct < 8; ct++)
            acc[ct] = __builtin_amdgcn_wmma_f32_16x16x32_bf16(false, Pf, false, Vf[ct], (short)0, acc[ct], false, false);
        acc[8] = __builtin_amdgcn_wmma_f32_16x16x32_bf16(false, Pf, false, onesf, (short)0, acc[8], false, false);
    }

    // epilogue: store bf16 partial O and f32 partial l (column 0 of acc[8])
    bf16*  po = pO + (size_t)split * BT * HDIM;
    float* pl = pL + (size_t)split * BT;
    #pragma unroll
    for (int r = 0; r < 8; r++) {
        int t = qbase + half * 8 + r;
        bf16* orow = po + (size_t)(b * SEQ + t) * HDIM;
        #pragma unroll
        for (int ct = 0; ct < 8; ct++) orow[ct * 16 + n] = (bf16)acc[ct][r];
        if (n == 0) pl[b * SEQ + t] = acc[8][r];
    }
}

// ---------------------------------------------------------------------------
// Kernel 3: combine the KSPLIT partials:  out = (O0+O1)/(l0+l1)
// ---------------------------------------------------------------------------
__global__ void combine_kernel(const bf16* __restrict__ pO, const float* __restrict__ pL,
                               float* __restrict__ out) {
    int idx = blockIdx.x * 256 + threadIdx.x;          // [0, BT*HDIM)
    int g   = idx >> 7;                                // row index (b*SEQ+t)
    float o = (float)pO[idx] + (float)pO[(size_t)BT * HDIM + idx];
    float l = pL[g] + pL[BT + g];
    out[idx] = o / l;
}

// ---------------------------------------------------------------------------
extern "C" void kernel_launch(void* const* d_in, const int* in_sizes, int n_in,
                              void* d_out, int out_size, void* d_ws, size_t ws_size,
                              hipStream_t stream) {
    const float* x  = (const float*)d_in[0];
    const float* Wk = (const float*)d_in[1];
    const float* Wq = (const float*)d_in[2];
    const float* Wv = (const float*)d_in[3];

    bf16* ws = (bf16*)d_ws;
    bf16* qb = ws;                                   // [BT][H]   bf16 (pre-scaled)
    bf16* kb = qb + (size_t)BT * HDIM;               // [BT][H]   bf16
    bf16* vt = kb + (size_t)BT * HDIM;               // [B][H][T] bf16 (transposed)
    bf16* wt = vt + (size_t)BT * HDIM;               // [3][H][C] bf16 (transposed)
    bf16* pO = wt + (size_t)3 * HDIM * CDIM;         // [KSPLIT][BT][H] bf16 partial O
    float* pL = (float*)(pO + (size_t)KSPLIT * BT * HDIM);  // [KSPLIT][BT] f32 partial l

    wconv_kernel<<<(3 * HDIM * CDIM) / 256, 256, 0, stream>>>(Wq, Wk, Wv, wt);
    qkv_kernel<<<dim3(BT / 16, 3), 64, 0, stream>>>(x, wt, qb, kb, vt);
    attn_kernel<<<dim3(SEQ / 64, BATCH, KSPLIT), 128, 0, stream>>>(qb, kb, vt, pO, pL);
    combine_kernel<<<(BT * HDIM) / 256, 256, 0, stream>>>(pO, pL, (float*)d_out);
}
